// MaskedCRF_86689619903491
// MI455X (gfx1250) — compile-verified, matrix operand order checked by reading, not run on hardware
//
#include <hip/hip_runtime.h>

#define NUM_TAGS 96
#define BATCH 512
#define SEQ 512

typedef __attribute__((ext_vector_type(16))) _Float16 v16h;
typedef __attribute__((ext_vector_type(8)))  float    v8f;

#define LOG2E 1.44269504f
#define LN2   0.69314718f

// DPP16 lane exchange, pure VALU (no LDS): ctrl is a DPP16 control code.
template <int CTRL>
__device__ __forceinline__ float dpp_mov(float x) {
    int xi = __builtin_bit_cast(int, x);
    int r  = __builtin_amdgcn_update_dpp(xi, xi, CTRL, 0xF, 0xF, false);
    return __builtin_bit_cast(float, r);
}
// Max-reduce across the 16 lanes of a half-wave (one C-tile row):
// xor1 = quad_perm[1,0,3,2]=0xB1, xor2 = quad_perm[2,3,0,1]=0x4E,
// then row_half_mirror (0x141) and row_mirror (0x140).
__device__ __forceinline__ float half16_max(float v) {
    v = fmaxf(v, dpp_mov<0xB1>(v));
    v = fmaxf(v, dpp_mov<0x4E>(v));
    v = fmaxf(v, dpp_mov<0x141>(v));
    v = fmaxf(v, dpp_mov<0x140>(v));
    return v;
}

// ---------------------------------------------------------------------------
__global__ void crf_zero_out(float* out) { out[0] = 0.0f; }

// ---------------------------------------------------------------------------
// Numerator: one thread per batch row (pure gather chain).
// ---------------------------------------------------------------------------
__global__ __launch_bounds__(256) void crf_numerator(
    const float* __restrict__ logits, const int* __restrict__ tags,
    const int* __restrict__ mask, const float* __restrict__ trans,
    const float* __restrict__ startT, const float* __restrict__ endT,
    float* __restrict__ out)
{
    int b = blockIdx.x * blockDim.x + threadIdx.x;
    if (b >= BATCH) return;

    const int*   tg = tags + (size_t)b * SEQ;
    const int*   mk = mask + (size_t)b * SEQ;
    const float* em = logits + (size_t)b * SEQ * NUM_TAGS;

    int prev = tg[0];
    float score = startT[prev] + em[prev];
    int cnt = (mk[0] > 0) ? 1 : 0;

    for (int t = 1; t < SEQ; ++t) {
        int cur = tg[t];
        if (mk[t] > 0) {
            score += trans[prev * NUM_TAGS + cur] + em[t * NUM_TAGS + cur];
            ++cnt;
        }
        prev = cur;  // reference uses raw tg[t-1] regardless of mask
    }
    score += endT[tg[cnt - 1]];
    atomicAdd(out, score);
}

// ---------------------------------------------------------------------------
// Denominator: forward algorithm in exp-space so the inner logsumexp becomes a
// WMMA GEMM.  One workgroup = 16 batch rows (WMMA M), 6 wave32 = 6 N-tiles.
//   p[i]      = exp(alpha[i] - rowmax)          (f16, in [0,1])
//   expT[i,j] = exp(trans[i,j])                 (f16, in [0.905,1.105])
//   alpha'[j] = rowmax + log( (p · expT)[j] ) + em[t,j]   (masked select)
// Row maxima for step t+1 are reduced in-register from the step-t C tile via
// a 4-step DPP16 butterfly (VALU-only).  2 barriers per step.
// ---------------------------------------------------------------------------
__global__ __launch_bounds__(192) void crf_denominator(
    const float* __restrict__ logits, const int* __restrict__ mask,
    const float* __restrict__ trans, const float* __restrict__ startT,
    const float* __restrict__ endT, float* __restrict__ out)
{
    __shared__ _Float16 expT[NUM_TAGS * NUM_TAGS];  // 18 KB
    __shared__ float    alpha[16 * NUM_TAGS];       // 6 KB
    __shared__ _Float16 pbuf[16 * NUM_TAGS];        // 3 KB
    __shared__ float    pm[16 * 6];                 // per-(row, wave) max partials
    __shared__ float    rowmax[16];
    __shared__ int      mkbuf[16];

    const int tid    = threadIdx.x;
    const int lane   = tid & 31;
    const int wave   = tid >> 5;          // 0..5 -> N-tile
    const int b_base = blockIdx.x * 16;
    const int n_base = wave * 16;
    const int laneHi = (lane >> 4);       // 0 for lanes 0-15, 1 for 16-31

    // Phase-A strip ownership: row rA, contiguous 8 columns starting at jA.
    const int rA = tid / 12;
    const int sA = tid - rA * 12;
    const int jA = sA * 8;

    // Stage exp(transitions) into LDS as f16 (time-invariant B matrix).
    for (int i = tid; i < NUM_TAGS * NUM_TAGS; i += 192)
        expT[i] = (_Float16)__builtin_amdgcn_exp2f(trans[i] * LOG2E);

    // alpha0 = start_transitions + em[t=0]
    for (int i = tid; i < 16 * NUM_TAGS; i += 192) {
        int r = i / NUM_TAGS, j = i - r * NUM_TAGS;
        int b = b_base + r;
        alpha[i] = startT[j] + logits[((size_t)b * SEQ) * NUM_TAGS + j];
    }
    __syncthreads();

    // Initial partial maxima of alpha0: 96 threads, one (row, 16-col segment).
    if (tid < 96) {
        int r = tid / 6, s = tid - r * 6;
        float m = -1e30f;
        for (int j = s * 16; j < s * 16 + 16; ++j)
            m = fmaxf(m, alpha[r * NUM_TAGS + j]);
        pm[r * 6 + s] = m;
    }

    // Preload this wave's three B tiles (32x16 f16) into registers for the
    // whole scan.  16-bit B 32x16 lane layout: N = lane&15,
    // element e holds K = e + 8*(lane>=16) + 8*(e>=8).
    const int Ncol = n_base + (lane & 15);
    v16h B0, B1, B2;
#pragma unroll
    for (int e = 0; e < 16; ++e) {
        int K = e + (laneHi << 3) + ((e >> 3) << 3);
        B0[e] = expT[(K)      * NUM_TAGS + Ncol];
        B1[e] = expT[(32 + K) * NUM_TAGS + Ncol];
        B2[e] = expT[(64 + K) * NUM_TAGS + Ncol];
    }
    __syncthreads();   // pm ready for first loop iteration

    const int Mrow = lane & 15;  // A layout: M = lane&15
    const int jC   = n_base + (lane & 15);

    for (int t = 1; t < SEQ; ++t) {
        // ---- Phase A: redundant 6-way pm reduce per strip; p = exp2(...) ----
        float m = pm[rA * 6 + 0];
        m = fmaxf(m, pm[rA * 6 + 1]);
        m = fmaxf(m, pm[rA * 6 + 2]);
        m = fmaxf(m, pm[rA * 6 + 3]);
        m = fmaxf(m, pm[rA * 6 + 4]);
        m = fmaxf(m, pm[rA * 6 + 5]);
        if (sA == 0) rowmax[rA] = m;
        if (sA == 1) mkbuf[rA] = mask[(size_t)(b_base + rA) * SEQ + t];
#pragma unroll
        for (int q = 0; q < 8; ++q) {
            float a = alpha[rA * NUM_TAGS + jA + q];
            pbuf[rA * NUM_TAGS + jA + q] =
                (_Float16)__builtin_amdgcn_exp2f((a - m) * LOG2E);
        }
        __syncthreads();   // pbuf/rowmax/mkbuf ready; pm reads done

        // ---- Phase B: C(16x16) += A(16x32) x B(32x16), 3 K-steps ----
        v8f c = {};
#pragma unroll
        for (int kk = 0; kk < 3; ++kk) {
            v16h a;
#pragma unroll
            for (int e = 0; e < 16; ++e) {
                int K = e + (laneHi << 3) + ((e >> 3) << 3);
                a[e] = pbuf[Mrow * NUM_TAGS + kk * 32 + K];
            }
            const v16h b = (kk == 0) ? B0 : (kk == 1) ? B1 : B2;
            c = __builtin_amdgcn_wmma_f32_16x16x32_f16(
                    /*neg_a=*/false, a, /*neg_b=*/false, b,
                    /*c_mod=*/(short)0, c, /*reuse_a=*/false, /*reuse_b=*/false);
        }

        // ---- alpha update + in-register partial max for next step ----
        // C/D layout: VGPR v -> M = v + 8*(lane>=16), N = lane&15.
        float red[8];
#pragma unroll
        for (int v = 0; v < 8; ++v) {
            int M = v + (laneHi << 3);
            int b = b_base + M;
            float em_v = logits[((size_t)b * SEQ + t) * NUM_TAGS + jC];
            float nxt  = rowmax[M] + LN2 * __builtin_amdgcn_logf(c[v]) + em_v;
            float old  = alpha[M * NUM_TAGS + jC];
            float sv   = (mkbuf[M] > 0) ? nxt : old;
            alpha[M * NUM_TAGS + jC] = sv;
            red[v] = half16_max(sv);   // DPP butterfly, VALU only
        }
        if ((lane & 15) == 0) {        // lane 0 (M=v) and lane 16 (M=v+8)
#pragma unroll
            for (int v = 0; v < 8; ++v)
                pm[(v + (laneHi << 3)) * 6 + wave] = red[v];
        }
        __syncthreads();   // alpha + pm ready for next iteration
    }

    // ---- denominator = logsumexp_j(alpha + end_transitions) per row ----
    if (tid < 16) {
        float m = -1e30f;
        for (int j = 0; j < NUM_TAGS; ++j)
            m = fmaxf(m, alpha[tid * NUM_TAGS + j] + endT[j]);
        float s = 0.0f;
        for (int j = 0; j < NUM_TAGS; ++j)
            s += __builtin_amdgcn_exp2f((alpha[tid * NUM_TAGS + j] + endT[j] - m) * LOG2E);
        float denom = m + LN2 * __builtin_amdgcn_logf(s);
        atomicAdd(out, -denom);
    }
}

// ---------------------------------------------------------------------------
extern "C" void kernel_launch(void* const* d_in, const int* in_sizes, int n_in,
                              void* d_out, int out_size, void* d_ws, size_t ws_size,
                              hipStream_t stream) {
    (void)in_sizes; (void)n_in; (void)out_size; (void)d_ws; (void)ws_size;

    const float* logits = (const float*)d_in[0];
    const int*   tags   = (const int*)d_in[1];
    const int*   mask   = (const int*)d_in[2];
    const float* trans  = (const float*)d_in[3];
    const float* startT = (const float*)d_in[4];
    const float* endT   = (const float*)d_in[5];
    float*       out    = (float*)d_out;

    crf_zero_out<<<1, 1, 0, stream>>>(out);
    crf_numerator<<<(BATCH + 255) / 256, 256, 0, stream>>>(
        logits, tags, mask, trans, startT, endT, out);
    crf_denominator<<<BATCH / 16, 192, 0, stream>>>(
        logits, mask, trans, startT, endT, out);
}
